// GEML_28595892257610
// MI455X (gfx1250) — matrix operand order repaired and checked by reading, not compiled, not run on hardware
//
#include <hip/hip_runtime.h>
#include <hip/hip_bf16.h>
#include <cstdint>
#include <cstddef>

typedef __attribute__((ext_vector_type(16))) _Float16 v16h;
typedef __attribute__((ext_vector_type(8)))  float    v8f;
typedef uint32_t u32x4 __attribute__((ext_vector_type(4)));
typedef uint32_t u32x8 __attribute__((ext_vector_type(8)));

#define T_STEPS 8
#define BATCH   16
#define NODES   512
#define DIM     384
#define BN      (BATCH*NODES)   // 8192
#define G4      (4*DIM)         // 1536
#define KCAT    (2*DIM)         // 768
#define NT_G    (G4/16)         // 96 column tiles of gate matrix
#define KT_CAT  (KCAT/32)       // 24 k-tiles over concatenated [x,h]
#define DT      (DIM/16)        // 24 d tiles
#define KT_D    (DIM/32)        // 12 k-tiles over D

__device__ __forceinline__ float sigf(float v){ return 1.0f/(1.0f+__expf(-v)); }

__device__ __forceinline__ v8f wmma32(v16h a, v16h b, v8f c){
  // D = A(16x32 f16) * B(32x16 f16) + C(16x16 f32)
  return __builtin_amdgcn_wmma_f32_16x16x32_f16(false, a, false, b, (short)0, c,
                                                false, false);
}

// Load a 16x32 f16 A-fragment; base points at this lane's row slice (M=lane&15)
// at the k-tile origin. Layout: VGPR v, lo lanes: K = 2v; hi lanes: +16.
// Pairs contiguous.
__device__ __forceinline__ v16h load_a_frag(const _Float16* base, int hi){
  v16h a;
#pragma unroll
  for (int v = 0; v < 8; ++v){
    int kk = 2*v + (hi << 4);
    union { uint32_t u; _Float16 h2[2]; } cv;
    cv.u = *(const uint32_t*)(base + kk);
    a[2*v]   = cv.h2[0];
    a[2*v+1] = cv.h2[1];
  }
  return a;
}

// ---------------------------------------------------------------------------
// 1) spat = sigmoid(x) -> f16 planes; mean over T -> f32
// ---------------------------------------------------------------------------
__global__ void k_sigmoid_pack(const float* __restrict__ x,
                               _Float16* __restrict__ xh,
                               float* __restrict__ meanb){
  size_t idx = (size_t)blockIdx.x*blockDim.x + threadIdx.x;
  if (idx >= (size_t)BN*DIM) return;
  float acc = 0.f;
#pragma unroll
  for (int t = 0; t < T_STEPS; ++t){
    float s = sigf(x[(size_t)t*BN*DIM + idx]);
    xh[(size_t)t*BN*DIM + idx] = (_Float16)s;
    acc += s;
  }
  meanb[idx] = acc * (1.0f/T_STEPS);
}

// ---------------------------------------------------------------------------
// 2) Pack [W_ih; W_hh] (as B matrix, K=768 x N=1536) into per-lane WMMA
//    B-fragment layout: Wp[((nt*KT_CAT+kt)*32 + lane)*16 + hh],
//    half hh <-> K = kt*32 + (lane>=16?16:0) + hh, N = nt*16 + lane%16.
// ---------------------------------------------------------------------------
__global__ void k_pack_w(const float* __restrict__ W_ih,
                         const float* __restrict__ W_hh,
                         _Float16* __restrict__ Wp){
  int idx = blockIdx.x*blockDim.x + threadIdx.x;
  if (idx >= NT_G*KT_CAT*512) return;
  int hh   = idx & 15;
  int lane = (idx >> 4) & 31;
  int kt   = (idx >> 9) % KT_CAT;
  int nt   = (idx >> 9) / KT_CAT;
  int K = kt*32 + ((lane >> 4) << 4) + hh;
  int n = nt*16 + (lane & 15);
  float v = (K < DIM) ? W_ih[(size_t)n*DIM + K] : W_hh[(size_t)n*DIM + (K - DIM)];
  Wp[idx] = (_Float16)v;
}

// Same packing for Wg.T (K=384 x N=384): B[k][n] = Wg[n][k]
__global__ void k_pack_wg(const float* __restrict__ Wg, _Float16* __restrict__ Wgp){
  int idx = blockIdx.x*blockDim.x + threadIdx.x;
  if (idx >= DT*KT_D*512) return;
  int hh   = idx & 15;
  int lane = (idx >> 4) & 31;
  int kt   = (idx >> 9) % KT_D;
  int nt   = (idx >> 9) / KT_D;
  int K = kt*32 + ((lane >> 4) << 4) + hh;
  int n = nt*16 + (lane & 15);
  Wgp[idx] = (_Float16)Wg[(size_t)n*DIM + K];
}

// ---------------------------------------------------------------------------
// 3) Fused LSTM. Per block: 16 rows. gates = [x_t, h] @ [W_ih;W_hh]^T via WMMA
//    (K=768, split 12 x-k-tiles + 12 h-k-tiles). x_t tile is DMA'd into LDS by
//    the Tensor Data Mover (tensor_load_to_lds, TENSORcnt); h round-trips via
//    LDS. Each wave owns 3 d-tiles x the 4 gate-aligned N-tiles so i/f/g/o of
//    an element share lane/VGPR slot -> c/h update is pure in-register VALU;
//    c persists in VGPRs across all 8 steps.
// ---------------------------------------------------------------------------
__global__ __launch_bounds__(256) void k_lstm(
    const _Float16* __restrict__ xh, const _Float16* __restrict__ Wp,
    const float* __restrict__ b_ih, const float* __restrict__ b_hh,
    const float* __restrict__ meanb, float* __restrict__ temp){
  __shared__ _Float16 s_x[16][DIM];   // sigmoid(x_t) tile (contiguous: TDM dest)
  __shared__ _Float16 s_h[16][DIM];   // recurrent h tile
  const int tid  = threadIdx.x;
  const int w    = tid >> 5;
  const int lane = tid & 31;
  const int hi   = lane >> 4;
  const int col  = lane & 15;
  const int row0 = blockIdx.x * 16;

  // zero h
  for (int i = tid; i < 16*(DIM/2); i += 256){
    int m = i / (DIM/2), p = i % (DIM/2);
    *(uint32_t*)&s_x[0][0]; // keep s_x referenced uniformly (no-op)
    *(uint32_t*)&s_h[m][2*p] = 0u;
  }

  // Tensor DMA descriptor group 1 (time-invariant): 2D tensor, 2-byte elems,
  // tile = 384 x 16, tensor_dim0 = 384 (stride 384), tensor_dim1 = T*BN rows.
  u32x8 g1;
  g1[0] = 1u << 16;                      // data_size = 2 bytes
  g1[1] = (uint32_t)DIM << 16;           // tensor_dim0[15:0] @ bits[63:48]
  g1[2] = 0u;                            // tensor_dim0 hi; tensor_dim1 lo16 (65536 -> 0)
  g1[3] = ((uint32_t)DIM << 16) | 1u;    // tensor_dim1 hi = 1; tile_dim0 = 384
  g1[4] = 16u;                           // tile_dim1 = 16 (tile_dim2 = 0)
  g1[5] = (uint32_t)DIM;                 // tensor_dim0_stride = 384
  g1[6] = 0u;
  g1[7] = 0u;
  const uint32_t lds_x = (uint32_t)(uintptr_t)(&s_x[0][0]);

  v8f   creg[3];
  float bsum[3][4];
  float hreg[3][8];
#pragma unroll
  for (int q = 0; q < 3; ++q){
#pragma unroll
    for (int j = 0; j < 8; ++j) creg[q][j] = 0.f;
    int dcol = (w + 8*q)*16 + col;
#pragma unroll
    for (int g = 0; g < 4; ++g)
      bsum[q][g] = b_ih[g*DIM + dcol] + b_hh[g*DIM + dcol];
  }

  for (int t = 0; t < T_STEPS; ++t){
    // --- TDM: DMA the 16x384 f16 x_t tile global -> LDS (wave 0 issues) ---
    if (w == 0){
      uint64_t ga = (uint64_t)(uintptr_t)(xh + ((size_t)t*BN + row0)*DIM);
      u32x4 g0;
      g0[0] = 1u;                                   // count = 1 (valid D#)
      g0[1] = lds_x;                                // LDS dest address
      g0[2] = (uint32_t)ga;                         // global_addr[31:0]
      g0[3] = (uint32_t)(ga >> 32) | (2u << 30);    // global_addr[56:32] | type=2
      asm volatile("tensor_load_to_lds %0, %1" :: "s"(g0), "s"(g1) : "memory");
      __builtin_amdgcn_s_wait_tensorcnt(0);
    }
    __syncthreads();

#pragma unroll
    for (int q = 0; q < 3; ++q){
      const int dt = w + 8*q;
      v8f acc[4];
#pragma unroll
      for (int g = 0; g < 4; ++g)
#pragma unroll
        for (int j = 0; j < 8; ++j) acc[g][j] = 0.f;

      const _Float16* rowx = &s_x[lane & 15][0];
      const _Float16* rowh = &s_h[lane & 15][0];
      // K-tiles 0..11: A from x region
      for (int kt = 0; kt < KT_D; ++kt){
        v16h a = load_a_frag(rowx + kt*32, hi);
        const _Float16* wpb =
            Wp + ((size_t)(dt*KT_CAT + kt)*32 + lane)*16;
        __builtin_prefetch(wpb + 512, 0, 0);
#pragma unroll
        for (int g = 0; g < 4; ++g){
          v16h b = *(const v16h*)(wpb + (size_t)g*DT*KT_CAT*512);
          acc[g] = wmma32(a, b, acc[g]);
        }
      }
      // K-tiles 12..23: A from h region
      for (int kt = 0; kt < KT_D; ++kt){
        v16h a = load_a_frag(rowh + kt*32, hi);
        const _Float16* wpb =
            Wp + ((size_t)(dt*KT_CAT + (kt + KT_D))*32 + lane)*16;
        __builtin_prefetch(wpb + 512, 0, 0);
#pragma unroll
        for (int g = 0; g < 4; ++g){
          v16h b = *(const v16h*)(wpb + (size_t)g*DT*KT_CAT*512);
          acc[g] = wmma32(a, b, acc[g]);
        }
      }
#pragma unroll
      for (int j = 0; j < 8; ++j){
        float iv = sigf (acc[0][j] + bsum[q][0]);
        float fv = sigf (acc[1][j] + bsum[q][1]);
        float gv = tanhf(acc[2][j] + bsum[q][2]);
        float ov = sigf (acc[3][j] + bsum[q][3]);
        float cc = fv*creg[q][j] + iv*gv;
        creg[q][j] = cc;
        hreg[q][j] = ov * tanhf(cc);
      }
    }
    __syncthreads();   // all GEMM reads done before h / s_x are overwritten

    if (t < T_STEPS - 1){
#pragma unroll
      for (int q = 0; q < 3; ++q){
        int d = (w + 8*q)*16 + col;
#pragma unroll
        for (int j = 0; j < 8; ++j)
          s_h[j + 8*hi][d] = (_Float16)hreg[q][j];
      }
    } else {
#pragma unroll
      for (int q = 0; q < 3; ++q){
        int d = (w + 8*q)*16 + col;
#pragma unroll
        for (int j = 0; j < 8; ++j){
          int m = j + 8*hi;
          size_t o = (size_t)(row0 + m)*DIM + d;
          temp[o] = hreg[q][j] + meanb[o];   // residual with temporal mean
        }
      }
    }
  }
}

// ---------------------------------------------------------------------------
// 4) BatchNorm stats: one block per channel
// ---------------------------------------------------------------------------
__global__ void k_bn_stats(const float* __restrict__ temp, float* __restrict__ stats){
  __shared__ float s1[256], s2[256];
  int ch = blockIdx.x, tid = threadIdx.x;
  float sum = 0.f, sq = 0.f;
  for (int r = tid; r < BN; r += 256){
    float v = temp[(size_t)r*DIM + ch];
    sum += v; sq += v*v;
  }
  s1[tid] = sum; s2[tid] = sq;
  __syncthreads();
  for (int s = 128; s > 0; s >>= 1){
    if (tid < s){ s1[tid] += s1[tid+s]; s2[tid] += s2[tid+s]; }
    __syncthreads();
  }
  if (tid == 0){
    float mu  = s1[0]*(1.0f/BN);
    float var = s2[0]*(1.0f/BN) - mu*mu;
    stats[ch]       = mu;
    stats[DIM + ch] = rsqrtf(var + 1e-5f);
  }
}

__global__ void k_bn_apply(float* __restrict__ temp, _Float16* __restrict__ th,
                           const float* __restrict__ stats,
                           const float* __restrict__ gamma,
                           const float* __restrict__ beta){
  size_t idx = (size_t)blockIdx.x*blockDim.x + threadIdx.x;
  if (idx >= (size_t)BN*DIM) return;
  int d = (int)(idx % DIM);
  float v = (temp[idx] - stats[d]) * stats[DIM + d] * gamma[d] + beta[d];
  temp[idx] = v;
  th[idx]   = (_Float16)v;
}

// ---------------------------------------------------------------------------
// 5) res_D[row] = temp[row] . Wd + bd
// ---------------------------------------------------------------------------
__global__ void k_resd(const float* __restrict__ temp, const float* __restrict__ Wd,
                       const float* __restrict__ bd, float* __restrict__ outD){
  __shared__ float red[64];
  int row = blockIdx.x, tid = threadIdx.x;
  float s = 0.f;
  for (int k = tid; k < DIM; k += 64) s += temp[(size_t)row*DIM + k]*Wd[k];
  red[tid] = s;
  __syncthreads();
  for (int st = 32; st > 0; st >>= 1){
    if (tid < st) red[tid] += red[tid + st];
    __syncthreads();
  }
  if (tid == 0) outD[row] = red[0] + bd[0];
}

// ---------------------------------------------------------------------------
// 6) A2 = temp @ Wg.T + bg  (WMMA, f16 out) -- M=8192, N=K=384.
//    32-row A tile staged via async global->LDS loads (ASYNCcnt path).
// ---------------------------------------------------------------------------
__global__ __launch_bounds__(256) void k_gemm_wg(
    const _Float16* __restrict__ th, const _Float16* __restrict__ Wgp,
    const float* __restrict__ bgv, _Float16* __restrict__ a2h){
  __shared__ _Float16 s_a[32][DIM];    // contiguous 24 KB == global tile
  const int tid  = threadIdx.x;
  const int w    = tid >> 5;
  const int lane = tid & 31;
  const int hi   = lane >> 4;
  const int col  = lane & 15;
  const int row0 = blockIdx.x * 32;

  // async DMA stage: 1536 x 16B chunks, 6 per thread
  {
    const uint32_t lds0 = (uint32_t)(uintptr_t)(&s_a[0][0]);
    const char* srcb = (const char*)(th + (size_t)row0*DIM);
#pragma unroll
    for (int c = 0; c < 6; ++c){
      int i = tid + c*256;
      uint32_t ldso = lds0 + (uint32_t)i*16u;
      uint64_t ga   = (uint64_t)(uintptr_t)(srcb + (size_t)i*16);
      asm volatile("global_load_async_to_lds_b128 %0, %1, off"
                   :: "v"(ldso), "v"(ga) : "memory");
    }
    asm volatile("s_wait_asynccnt 0x0" ::: "memory");
  }
  __syncthreads();

  const int mt  = w & 1;          // 2 m-tiles
  const int ng0 = (w >> 1)*6;     // 24 n-tiles / 4 wave-groups
  const _Float16* rowp = &s_a[mt*16 + (lane & 15)][0];

  for (int nn = 0; nn < 6; ++nn){
    int nt = ng0 + nn;
    v8f acc;
#pragma unroll
    for (int j = 0; j < 8; ++j) acc[j] = 0.f;
    for (int kt = 0; kt < KT_D; ++kt){
      v16h a = load_a_frag(rowp + kt*32, hi);
      v16h b = *(const v16h*)(Wgp + ((size_t)(nt*KT_D + kt)*32 + lane)*16);
      acc = wmma32(a, b, acc);
    }
    int n = nt*16 + col;
    float bb = bgv[n];
#pragma unroll
    for (int j = 0; j < 8; ++j){
      int m = mt*16 + j + 8*hi;
      a2h[(size_t)(row0 + m)*DIM + n] = (_Float16)(acc[j] + bb);
    }
  }
}

// ---------------------------------------------------------------------------
// 7) res_G[b] = A2[b] @ temp[b].T  (WMMA; B-fragment = 32 contiguous bytes of
//    a th row, since B[k][n] = th[b,n,k]) -- M=N=512, K=384, 16 batches
// ---------------------------------------------------------------------------
__global__ __launch_bounds__(256) void k_gemm_g(
    const _Float16* __restrict__ a2h, const _Float16* __restrict__ th,
    float* __restrict__ outG){
  const int b   = blockIdx.z;
  const int i0  = blockIdx.x * 64;
  const int j0  = blockIdx.y * 64;
  const int tid = threadIdx.x;
  const int w   = tid >> 5, lane = tid & 31, hi = lane >> 4, col = lane & 15;
  const int mt  = w >> 1;          // 4 m-tiles
  const int ntb = (w & 1)*2;       // 2 n-tiles per wave
  const size_t rowbase = (size_t)b*NODES;

  const _Float16* arow = a2h + (rowbase + i0 + mt*16 + (lane & 15))*(size_t)DIM;
  v8f acc[2];
#pragma unroll
  for (int jj = 0; jj < 2; ++jj)
#pragma unroll
    for (int j = 0; j < 8; ++j) acc[jj][j] = 0.f;

  for (int kt = 0; kt < KT_D; ++kt){
    v16h a = load_a_frag(arow + kt*32, hi);
#pragma unroll
    for (int jj = 0; jj < 2; ++jj){
      int n = j0 + (ntb + jj)*16 + col;
      const _Float16* bp = th + (rowbase + n)*(size_t)DIM + kt*32 + hi*16;
      v16h bfr = *(const v16h*)bp;
      acc[jj] = wmma32(a, bfr, acc[jj]);
    }
  }
  float* out = outG + (size_t)b*NODES*NODES;
#pragma unroll
  for (int jj = 0; jj < 2; ++jj)
#pragma unroll
    for (int j = 0; j < 8; ++j){
      int i = i0 + mt*16 + j + 8*hi;
      int n = j0 + (ntb + jj)*16 + col;
      out[(size_t)i*NODES + n] = acc[jj][j];
    }
}

// ---------------------------------------------------------------------------
extern "C" void kernel_launch(void* const* d_in, const int* in_sizes, int n_in,
                              void* d_out, int out_size, void* d_ws, size_t ws_size,
                              hipStream_t stream) {
  (void)in_sizes; (void)n_in; (void)out_size; (void)ws_size;
  const float* x    = (const float*)d_in[0];
  const float* W_ih = (const float*)d_in[1];
  const float* W_hh = (const float*)d_in[2];
  const float* b_ih = (const float*)d_in[3];
  const float* b_hh = (const float*)d_in[4];
  const float* gam  = (const float*)d_in[5];
  const float* bet  = (const float*)d_in[6];
  const float* Wd   = (const float*)d_in[7];
  const float* bd   = (const float*)d_in[8];
  const float* Wg   = (const float*)d_in[9];
  const float* bgv  = (const float*)d_in[10];
  float* outp = (float*)d_out;

  char* ws = (char*)d_ws;
  _Float16* xh    = (_Float16*)(ws + 0);           // 50,331,648 B
  float*    meanb = (float*)   (ws + 50331648);    // 12,582,912 B
  float*    temp  = (float*)   (ws + 62914560);    // 12,582,912 B
  _Float16* Wp    = (_Float16*)(ws + 75497472);    //  2,359,296 B
  _Float16* th    = (_Float16*)(ws + 77856768);    //  6,291,456 B
  _Float16* Wgp   = (_Float16*)(ws + 84148224);    //    294,912 B
  _Float16* a2h   = (_Float16*)(ws + 84443136);    //  6,291,456 B
  float*    stats = (float*)   (ws + 90734592);    //      3,072 B

  k_sigmoid_pack<<<(BN*DIM + 255)/256, 256, 0, stream>>>(x, xh, meanb);
  k_pack_w      <<<(NT_G*KT_CAT*512 + 255)/256, 256, 0, stream>>>(W_ih, W_hh, Wp);
  k_pack_wg     <<<(DT*KT_D*512 + 255)/256, 256, 0, stream>>>(Wg, Wgp);
  k_lstm        <<<BN/16, 256, 0, stream>>>(xh, Wp, b_ih, b_hh, meanb, temp);
  k_bn_stats    <<<DIM, 256, 0, stream>>>(temp, stats);
  k_bn_apply    <<<(BN*DIM + 255)/256, 256, 0, stream>>>(temp, th, stats, gam, bet);
  k_resd        <<<BN, 64, 0, stream>>>(temp, Wd, bd, outp);
  k_gemm_wg     <<<BN/32, 256, 0, stream>>>(th, Wgp, bgv, a2h);
  k_gemm_g      <<<dim3(NODES/64, NODES/64, BATCH), 256, 0, stream>>>(a2h, th, outp + BN);
}
